// ScaledDotProductAttention_48799418417366
// MI455X (gfx1250) — matrix operand order repaired and checked by reading, not compiled, not run on hardware
//
#include <hip/hip_runtime.h>
#include <hip/hip_bf16.h>

typedef __attribute__((ext_vector_type(16))) __bf16 v16bf;
typedef __attribute__((ext_vector_type(8)))  __bf16 v8bf;
typedef __attribute__((ext_vector_type(4)))  __bf16 v4bf;
typedef __attribute__((ext_vector_type(8)))  float  v8f;

#define B_DIM 4
#define H_DIM 8
#define S_DIM 2048
#define D_HEAD 64
#define BQ 128     // query rows per block (16 per wave)
#define TK 64      // key tile
#define NW 8       // waves per block
#define QTILES (S_DIM / BQ)   // 16 q-tiles per (b,h)
// scale = 1/sqrt(512) * log2(e): softmax done in base-2 (v_exp_f32 is exp2)
#define SCALE_LOG2 0.06375872022286571f
#define NEG_V  (-1.0e9f)

__device__ __forceinline__ __bf16 f2bf(float f) { return (__bf16)f; }

__device__ __forceinline__ float fast_exp2(float x) {
#if __has_builtin(__builtin_amdgcn_exp2f)
  return __builtin_amdgcn_exp2f(x);
#else
  return exp2f(x);
#endif
}

// DPP lane permute on the VALU (no LDS unit): ctrl is DPP16 control
template<int CTRL>
__device__ __forceinline__ float dpp_mov(float x) {
  int i = __builtin_bit_cast(int, x);
  i = __builtin_amdgcn_update_dpp(i, i, CTRL, 0xF, 0xF, true);
  return __builtin_bit_cast(float, i);
}
#define DPP_XOR1 0xB1  // quad_perm(1,0,3,2)
#define DPP_XOR2 0x4E  // quad_perm(2,3,0,1)
#define DPP_HM   0x141 // row_half_mirror (crosses quads within 8)
#define DPP_M    0x140 // row_mirror      (crosses 8-groups within 16)

// 16 contiguous bf16 from LDS (two 16B ds loads)
__device__ __forceinline__ v16bf lds16(const __bf16* p) {
  v8bf lo = *(const v8bf*)p;
  v8bf hi = *(const v8bf*)(p + 8);
  return __builtin_shufflevector(lo, hi, 0,1,2,3,4,5,6,7,8,9,10,11,12,13,14,15);
}

// A-layout gather: halves 0..7 at p0, halves 8..15 at p1
__device__ __forceinline__ v16bf lds16_split(const __bf16* p0, const __bf16* p1) {
  v8bf lo = *(const v8bf*)p0;
  v8bf hi = *(const v8bf*)p1;
  return __builtin_shufflevector(lo, hi, 0,1,2,3,4,5,6,7,8,9,10,11,12,13,14,15);
}

// One 64-key tile of flash attention. MASKED=true only for the final partial tile.
template<bool MASKED>
__device__ __forceinline__ void attn_tile(
    int k0, int len, int kend,
    const float* __restrict__ Kg, const float* __restrict__ Vg,
    __bf16* Klds, __bf16* Vtlds, __bf16* pw,
    const v16bf& aq0, const v16bf& aq1, const v16bf& bones,
    v8f (&Oacc)[4], float (&mrun)[8], float (&lrun)[8],
    int tid, int col, int hilo, int koffA, int kbB)
{
  if (k0 + TK < kend) {  // prefetch next K/V tile (global_prefetch_b8)
    __builtin_prefetch(&Kg[(long long)(k0 + TK) * D_HEAD + tid * 8], 0, 0);
    __builtin_prefetch(&Vg[(long long)(k0 + TK) * D_HEAD + tid * 8], 0, 0);
  }

  // ---- stage K (row-major) and V (transposed) tiles as bf16 in LDS ----
  __syncthreads();
  #pragma unroll
  for (int j = 0; j < 4; ++j) {
    int i4 = tid + j * (NW * 32);        // float4 index, 0..1023
    int kk = i4 >> 4;                    // key in tile (16 float4 per row)
    int dd = (i4 & 15) << 2;             // dim (multiple of 4)
    float4 kq = *(const float4*)&Kg[(long long)(k0 + kk) * D_HEAD + dd];
    float4 vq = *(const float4*)&Vg[(long long)(k0 + kk) * D_HEAD + dd];
    v4bf kp;
    kp[0] = f2bf(kq.x); kp[1] = f2bf(kq.y); kp[2] = f2bf(kq.z); kp[3] = f2bf(kq.w);
    *(v4bf*)&Klds[kk * D_HEAD + dd] = kp;     // one ds_store_b64
    Vtlds[(dd + 0) * TK + kk] = f2bf(vq.x);   // transposed scatter
    Vtlds[(dd + 1) * TK + kk] = f2bf(vq.y);
    Vtlds[(dd + 2) * TK + kk] = f2bf(vq.z);
    Vtlds[(dd + 3) * TK + kk] = f2bf(vq.w);
  }
  __syncthreads();

  // ---- scores: four 16-key subtiles, K-dim(d) = 64 via two WMMAs each ----
  v8f c[4];
  #pragma unroll
  for (int t = 0; t < 4; ++t) {
    const __bf16* kp = &Klds[(t * 16 + col) * D_HEAD];
    v16bf b0 = lds16_split(kp + koffA,      kp + 16 + koffA);
    v16bf b1 = lds16_split(kp + 32 + koffA, kp + 48 + koffA);
    v8f acc = (v8f){0.f,0.f,0.f,0.f,0.f,0.f,0.f,0.f};
    acc = __builtin_amdgcn_wmma_f32_16x16x32_bf16(false, aq0, false, b0, (short)0, acc, false, false);
    acc = __builtin_amdgcn_wmma_f32_16x16x32_bf16(false, aq1, false, b1, (short)0, acc, false, false);
    c[t] = acc;
  }

  // ---- key masking: only on the final partial tile ----
  if (MASKED) {
    #pragma unroll
    for (int t = 0; t < 4; ++t) {
      bool valid = (k0 + t * 16 + col) < len;
      #pragma unroll
      for (int r = 0; r < 8; ++r) c[t][r] = valid ? c[t][r] : NEG_V;
    }
  }

  // ---- online softmax: row max via DPP butterfly, then base-2 exp ----
  float corr[8];
  #pragma unroll
  for (int r = 0; r < 8; ++r) {
    float mx = fmaxf(fmaxf(c[0][r], c[1][r]), fmaxf(c[2][r], c[3][r]));
    mx = fmaxf(mx, dpp_mov<DPP_XOR1>(mx));
    mx = fmaxf(mx, dpp_mov<DPP_XOR2>(mx));
    mx = fmaxf(mx, dpp_mov<DPP_HM>(mx));
    mx = fmaxf(mx, dpp_mov<DPP_M>(mx));
    float newm = fmaxf(mrun[r], mx);
    corr[r] = fast_exp2(mrun[r] - newm);
    c[0][r] = fast_exp2(c[0][r] - newm);
    c[1][r] = fast_exp2(c[1][r] - newm);
    c[2][r] = fast_exp2(c[2][r] - newm);
    c[3][r] = fast_exp2(c[3][r] - newm);
    mrun[r] = newm;
  }
  #pragma unroll
  for (int r = 0; r < 8; ++r) {
    Oacc[0][r] *= corr[r]; Oacc[1][r] *= corr[r];
    Oacc[2][r] *= corr[r]; Oacc[3][r] *= corr[r];
  }

  // ---- C-layout -> A-layout for P via per-wave LDS round trip ----
  #pragma unroll
  for (int r = 0; r < 8; ++r) {
    int mm = r + 8 * hilo;
    #pragma unroll
    for (int t = 0; t < 4; ++t)
      pw[mm * TK + t * 16 + col] = f2bf(c[t][r]);
  }
  asm volatile("s_wait_dscnt 0" ::: "memory");  // wave-private region

  const __bf16* pr = pw + col * TK;
  v16bf ap0 = lds16_split(pr + koffA,      pr + 16 + koffA);  // keys  0..31
  v16bf ap1 = lds16_split(pr + 32 + koffA, pr + 48 + koffA);  // keys 32..63

  // ---- row sums of P via WMMA against all-ones B (broadcast in C layout) ----
  v8f csum = (v8f){0.f,0.f,0.f,0.f,0.f,0.f,0.f,0.f};
  csum = __builtin_amdgcn_wmma_f32_16x16x32_bf16(false, ap0, false, bones, (short)0, csum, false, false);
  csum = __builtin_amdgcn_wmma_f32_16x16x32_bf16(false, ap1, false, bones, (short)0, csum, false, false);
  #pragma unroll
  for (int r = 0; r < 8; ++r) lrun[r] = lrun[r] * corr[r] + csum[r];

  // ---- O += P * V  (K-dim = 64 keys; 4 output dim chunks) ----
  #pragma unroll
  for (int dc = 0; dc < 4; ++dc) {
    const __bf16* vr = &Vtlds[(dc * 16 + col) * TK];
    v16bf bv0 = lds16_split(vr + koffA,      vr + 16 + koffA);
    v16bf bv1 = lds16_split(vr + 32 + koffA, vr + 48 + koffA);
    Oacc[dc] = __builtin_amdgcn_wmma_f32_16x16x32_bf16(false, ap0, false, bv0, (short)0, Oacc[dc], false, false);
    Oacc[dc] = __builtin_amdgcn_wmma_f32_16x16x32_bf16(false, ap1, false, bv1, (short)0, Oacc[dc], false, false);
  }
}

__global__ __launch_bounds__(NW * 32)
void sdpa_fa_kernel(const float* __restrict__ Q, const float* __restrict__ K,
                    const float* __restrict__ V, const int* __restrict__ lens,
                    float* __restrict__ out)
{
  __shared__ __bf16 Klds[TK * D_HEAD];       // [key][d]       8 KB
  __shared__ __bf16 Vtlds[D_HEAD * TK];      // [d][key]       8 KB
  __shared__ __bf16 Plds[NW][16 * TK];       // per-wave P    16 KB

  const int tid  = threadIdx.x;
  const int lane = tid & 31;
  const int wv   = tid >> 5;

  const int tileId = blockIdx.x;          // 0..511
  const int qt = tileId & (QTILES - 1);   // q tile within (b,h)
  const int bh = tileId / QTILES;         // b*H + h
  const int b  = bh >> 3;

  int len = lens[b];
  if (len > S_DIM) len = S_DIM;
  const long long base = (long long)bh * (long long)(S_DIM * D_HEAD);
  const float* Kg = K + base;
  const float* Vg = V + base;

  const int col   = lane & 15;            // matrix column / row index
  const int hilo  = lane >> 4;            // lane half
  const int koffA = hilo * 8;             // A-operand K offset
  const int kbB   = hilo * 16;            // B-operand K offset

  // ---- Load Q A-operands once (bf16, scale*log2e folded in) ----
  const int mrow = qt * BQ + wv * 16 + col;
  const float* qrow = Q + base + (long long)mrow * D_HEAD;
  v16bf aq0, aq1;
  #pragma unroll
  for (int i = 0; i < 8; ++i) {
    aq0[i]     = f2bf(qrow[koffA + i]      * SCALE_LOG2);
    aq0[8 + i] = f2bf(qrow[16 + koffA + i] * SCALE_LOG2);
    aq1[i]     = f2bf(qrow[32 + koffA + i] * SCALE_LOG2);
    aq1[8 + i] = f2bf(qrow[48 + koffA + i] * SCALE_LOG2);
  }

  // all-ones B operand (for WMMA row sums)
  v16bf bones;
  #pragma unroll
  for (int i = 0; i < 16; ++i) bones[i] = f2bf(1.0f);

  v8f Oacc[4];
  #pragma unroll
  for (int dc = 0; dc < 4; ++dc) Oacc[dc] = (v8f){0.f,0.f,0.f,0.f,0.f,0.f,0.f,0.f};
  float mrun[8], lrun[8];
  #pragma unroll
  for (int r = 0; r < 8; ++r) { mrun[r] = -1.0e30f; lrun[r] = 0.f; }

  // full (unmasked) tiles, then at most one masked partial tile
  const int kfull = len & ~(TK - 1);
  const int kend  = (len + TK - 1) & ~(TK - 1);
  __bf16* pw = &Plds[wv][0];

  for (int k0 = 0; k0 < kfull; k0 += TK)
    attn_tile<false>(k0, len, kend, Kg, Vg, Klds, Vtlds, pw,
                     aq0, aq1, bones, Oacc, mrun, lrun,
                     tid, col, hilo, koffA, kbB);
  if (len & (TK - 1))
    attn_tile<true>(kfull, len, kend, Kg, Vg, Klds, Vtlds, pw,
                    aq0, aq1, bones, Oacc, mrun, lrun,
                    tid, col, hilo, koffA, kbB);

  // ---- epilogue: normalize and store ----
  #pragma unroll
  for (int r = 0; r < 8; ++r) {
    float inv = 1.0f / lrun[r];
    int mm = qt * BQ + wv * 16 + r + 8 * hilo;
    float* op = out + base + (long long)mm * D_HEAD + col;
    op[0]  = Oacc[0][r] * inv;
    op[16] = Oacc[1][r] * inv;
    op[32] = Oacc[2][r] * inv;
    op[48] = Oacc[3][r] * inv;
  }
}

extern "C" void kernel_launch(void* const* d_in, const int* in_sizes, int n_in,
                              void* d_out, int out_size, void* d_ws, size_t ws_size,
                              hipStream_t stream) {
  (void)in_sizes; (void)n_in; (void)out_size; (void)d_ws; (void)ws_size;
  const float* Q   = (const float*)d_in[0];
  const float* K   = (const float*)d_in[1];
  const float* V   = (const float*)d_in[2];
  const int* lens  = (const int*)d_in[3];
  float* out = (float*)d_out;

  dim3 grid(B_DIM * H_DIM * QTILES);         // 512 blocks
  dim3 block(NW * 32);                       // 8 wave32s
  sdpa_fa_kernel<<<grid, block, 0, stream>>>(Q, K, V, lens, out);
}
